// Encoder_11261404250721
// MI455X (gfx1250) — matrix operand order repaired and checked by reading, not compiled, not run on hardware
//
#include <hip/hip_runtime.h>

// ---------------------------------------------------------------------------
// Types for CDNA5 WMMA (wave32, 16x16x32 bf16 -> f32)
// ---------------------------------------------------------------------------
typedef __attribute__((ext_vector_type(16))) __bf16 v16bf;
typedef __attribute__((ext_vector_type(8)))  __bf16 v8bf;
typedef __attribute__((ext_vector_type(4)))  __bf16 bf16x4;
typedef __attribute__((ext_vector_type(8)))  float  v8f;
typedef __attribute__((ext_vector_type(4)))  float  f32x4;

// A-matrix fragment (16x32, 16-bit), per CDNA5 ISA A layout:
//   lanes 0-15  : row m=lane,    halves = K[0..7]  then K[16..23]
//   lanes 16-31 : row m=lane-16, halves = K[8..15] then K[24..31]
// Caller passes &tile[row*stride + ka] with ka = (lane<16) ? 0 : 8.
__device__ __forceinline__ v16bf ldfragA(const __bf16* p) {
  union { v16bf v; v8bf h[2]; } u;
  u.h[0] = *(const v8bf*)(p);        // K base .. base+7
  u.h[1] = *(const v8bf*)(p + 16);   // K base+16 .. base+23
  return u.v;
}

// B-matrix fragment (32x16, 16-bit), per CDNA5 ISA B layout (from the
// SWMMAC dense-B tables): same K order as A:
//   lanes 0-15  : col n=lane,    halves = K[0..7]  then K[16..23]
//   lanes 16-31 : col n=lane-16, halves = K[8..15] then K[24..31]
// Caller passes &tile[col*stride + kb] with kb = (lane<16) ? 0 : 8,
// where tile is B^T (i.e. [n][k] row-major).
__device__ __forceinline__ v16bf ldfragB(const __bf16* p) {
  union { v16bf v; v8bf h[2]; } u;
  u.h[0] = *(const v8bf*)(p);        // K base .. base+7
  u.h[1] = *(const v8bf*)(p + 16);   // K base+16 .. base+23
  return u.v;
}

__device__ __forceinline__ v8f wmma_bf16(v16bf a, v16bf b, v8f c) {
  return __builtin_amdgcn_wmma_f32_16x16x32_bf16(false, a, false, b,
                                                 (short)0, c, false, false);
}

// ---------------------------------------------------------------------------
// Weight transpose + f32 -> bf16 convert:  dst[z][n][k] = (bf16) src[z][k][n]
// grid: (N/32, K/32, L), block: (32, 8)
// ---------------------------------------------------------------------------
__global__ __launch_bounds__(256) void convT_kernel(
    const float* __restrict__ src, __bf16* __restrict__ dst, int K, int N) {
  __shared__ float tile[32][33];
  const size_t zo = (size_t)blockIdx.z * K * N;
  const int kb = blockIdx.y * 32, nb = blockIdx.x * 32;
  const int tx = threadIdx.x, ty = threadIdx.y;
#pragma unroll
  for (int i = 0; i < 32; i += 8)
    tile[ty + i][tx] = src[zo + (size_t)(kb + ty + i) * N + nb + tx];
  __syncthreads();
#pragma unroll
  for (int i = 0; i < 32; i += 8)
    dst[zo + (size_t)(nb + ty + i) * K + kb + tx] = (__bf16)tile[tx][ty + i];
}

// ---------------------------------------------------------------------------
// x[m,d] = tok_emb[src[m]][d] * sqrt(D) + PE(s, d), m = b*S + s
// grid: M blocks, 256 threads
// ---------------------------------------------------------------------------
__global__ __launch_bounds__(256) void embed_pos_kernel(
    const int* __restrict__ src, const float* __restrict__ emb,
    float* __restrict__ x) {
  const int D = 512, S = 2048;
  const int m = blockIdx.x;
  const int s = m & (S - 1);
  const int tok = src[m];
  const float sc = 22.627416997969522f;  // sqrt(512)
#pragma unroll
  for (int j = 0; j < 2; ++j) {
    int d = threadIdx.x + j * 256;
    int i = d >> 1;
    float freq = expf((float)(2 * i) * (-9.210340371976184f / (float)D));
    float ang = (float)s * freq;
    float pe = (d & 1) ? cosf(ang) : sinf(ang);
    x[(size_t)m * D + d] = emb[(size_t)tok * D + d] * sc + pe;
  }
}

// ---------------------------------------------------------------------------
// C[M,N] = act(A[M,K] @ Bt[N,K]^T + bias[N]).  A f32 (converted to bf16 in
// LDS), Bt pre-converted bf16 row-major [N][K].  Block tile 128x64, 8 waves,
// wave tile 32x32 (2x2 WMMA), K chunks of 32.
// grid: (N/64, M/128), block: 256
// ---------------------------------------------------------------------------
#define G_BM 128
#define G_BN 64
#define G_BK 32
#define G_LDA 40   // halves per LDS row (80B, 16B aligned, conflict-free)

__global__ __launch_bounds__(256) void gemm_bf16_kernel(
    const float* __restrict__ A, const __bf16* __restrict__ Bt,
    const float* __restrict__ bias, float* __restrict__ C,
    int N, int K, int relu) {
  __shared__ __bf16 sA[G_BM * G_LDA];
  __shared__ __bf16 sB[G_BN * G_LDA];

  const int t = threadIdx.x;
  const int lane = t & 31, w = t >> 5;
  const int wm = w & 3, wn = w >> 2;                // 4x2 wave grid
  const int bm = blockIdx.y * G_BM, bn = blockIdx.x * G_BN;

  v8f c[2][2];
  c[0][0] = {}; c[0][1] = {}; c[1][0] = {}; c[1][1] = {};

  const int lrow = lane & 15;
  const int ka = (lane < 16) ? 0 : 8;    // A-fragment K offset
  const int kb = (lane < 16) ? 0 : 8;    // B-fragment K offset

  const int ra = t >> 3, ca = (t & 7) * 4;   // sA fill coords
  const int rb = t >> 2, cbb = (t & 3) * 8;  // sB fill coords

  for (int k0 = 0; k0 < K; k0 += G_BK) {
    // ---- issue all global loads first (keeps them in flight together) ----
    f32x4 av[4];
#pragma unroll
    for (int i = 0; i < 4; ++i)
      av[i] = *(const f32x4*)&A[(size_t)(bm + ra + i * 32) * K + k0 + ca];
    v8bf bv8 = *(const v8bf*)&Bt[(size_t)(bn + rb) * K + k0 + cbb];
    // speculative prefetch of next K-chunk (silently dropped if OOB)
    __builtin_prefetch(&A[(size_t)(bm + ra) * K + k0 + G_BK + ca], 0, 1);
    __builtin_prefetch(&Bt[(size_t)(bn + rb) * K + k0 + G_BK + cbb], 0, 1);
    // ---- convert + stage into LDS ----
#pragma unroll
    for (int i = 0; i < 4; ++i)
      *(bf16x4*)&sA[(ra + i * 32) * G_LDA + ca] =
          __builtin_convertvector(av[i], bf16x4);
    *(v8bf*)&sB[rb * G_LDA + cbb] = bv8;
    __syncthreads();

    v16bf a0 = ldfragA(&sA[(wm * 32 + lrow) * G_LDA + ka]);
    v16bf a1 = ldfragA(&sA[(wm * 32 + 16 + lrow) * G_LDA + ka]);
    v16bf b0 = ldfragB(&sB[(wn * 32 + lrow) * G_LDA + kb]);
    v16bf b1 = ldfragB(&sB[(wn * 32 + 16 + lrow) * G_LDA + kb]);

    c[0][0] = wmma_bf16(a0, b0, c[0][0]);
    c[0][1] = wmma_bf16(a0, b1, c[0][1]);
    c[1][0] = wmma_bf16(a1, b0, c[1][0]);
    c[1][1] = wmma_bf16(a1, b1, c[1][1]);
    __syncthreads();
  }

  // Epilogue. C layout: vgpr r -> row rbase+r, lane -> col (lane&15).
  const int col = lane & 15;
  const int rbase = (lane < 16) ? 0 : 8;
#pragma unroll
  for (int ti = 0; ti < 2; ++ti)
#pragma unroll
    for (int tj = 0; tj < 2; ++tj) {
      int gm = bm + wm * 32 + ti * 16 + rbase;
      int gn = bn + wn * 32 + tj * 16 + col;
      float bv = bias[gn];
#pragma unroll
      for (int r = 0; r < 8; ++r) {
        float v = c[ti][tj][r] + bv;
        if (relu) v = fmaxf(v, 0.0f);
        C[(size_t)(gm + r) * N + gn] = v;
      }
    }
}

// ---------------------------------------------------------------------------
// Flash attention: per (b, h, 64 q rows) workgroup; 4 waves x 16 q rows.
// KV tiles of 64.  QK^T and P@V via WMMA bf16, online softmax in f32.
// Q and P are A operands (ldfragA); K-tile and V^T are B operands (ldfragB:
// for QK^T, B[k][n] = K[n][k] so the row-major K tile IS B^T; for P@V,
// B[k][n] = V[k][n] so B^T = V^T, staged transposed in LDS).
// grid: (S/64, H, B), block: 128
// ---------------------------------------------------------------------------
#define F_LD 72   // halves per LDS row for 64-wide tiles (144B, 16B aligned)

__global__ __launch_bounds__(128) void flash_attn_kernel(
    const float* __restrict__ q, const float* __restrict__ k,
    const float* __restrict__ v, const int* __restrict__ mask,
    float* __restrict__ o) {
  const int S = 2048, D = 512, HD = 64;
  __shared__ __bf16 sQ[64 * F_LD];
  __shared__ __bf16 sK[64 * F_LD];
  __shared__ __bf16 sV[64 * F_LD];          // V^T: [hd][kv]
  __shared__ __bf16 sP[4][16 * F_LD];       // per-wave P tile

  const int t = threadIdx.x, lane = t & 31, w = t >> 5;
  const int qb = blockIdx.x * 64;
  const int h = blockIdx.y, b = blockIdx.z;
  const size_t rowstart = (size_t)b * S * D + (size_t)h * HD;
  const int* mrow_ptr = mask + (size_t)b * S;

  const int fr = t >> 1, fc = (t & 1) * 32;  // fill coords: row, col base

  // load Q block (64 x 64) -> bf16 LDS
  {
    const float* src = q + rowstart + (size_t)(qb + fr) * D + fc;
    f32x4 tmp[8];
#pragma unroll
    for (int i = 0; i < 8; ++i) tmp[i] = *(const f32x4*)(src + i * 4);
#pragma unroll
    for (int i = 0; i < 8; ++i)
      *(bf16x4*)&sQ[fr * F_LD + fc + i * 4] =
          __builtin_convertvector(tmp[i], bf16x4);
  }
  __syncthreads();

  const int lrow = lane & 15;
  const int ka = (lane < 16) ? 0 : 8;
  const int kb = (lane < 16) ? 0 : 8;
  v16bf qf0 = ldfragA(&sQ[(w * 16 + lrow) * F_LD + ka]);        // K 0..31
  v16bf qf1 = ldfragA(&sQ[(w * 16 + lrow) * F_LD + 32 + ka]);   // K 32..63

  float mrun[8], lrun[8];
  v8f acc[4];
#pragma unroll
  for (int r = 0; r < 8; ++r) { mrun[r] = -3e38f; lrun[r] = 0.0f; }
  acc[0] = {}; acc[1] = {}; acc[2] = {}; acc[3] = {};
  const float scale = 0.125f;  // 1/sqrt(64)
  const int rbase = (lane < 16) ? 0 : 8;

  for (int kv = 0; kv < S; kv += 64) {
    __syncthreads();  // previous tile's sK/sV reads done
    {  // fill sK row-major, sV transposed; batch loads before stores
      const float* ksrc = k + rowstart + (size_t)(kv + fr) * D + fc;
      const float* vsrc = v + rowstart + (size_t)(kv + fr) * D + fc;
      f32x4 kt[8], vt[8];
#pragma unroll
      for (int i = 0; i < 8; ++i) kt[i] = *(const f32x4*)(ksrc + i * 4);
#pragma unroll
      for (int i = 0; i < 8; ++i) vt[i] = *(const f32x4*)(vsrc + i * 4);
      __builtin_prefetch(ksrc + (size_t)64 * D, 0, 1);  // next KV tile
      __builtin_prefetch(vsrc + (size_t)64 * D, 0, 1);
#pragma unroll
      for (int i = 0; i < 8; ++i)
        *(bf16x4*)&sK[fr * F_LD + fc + i * 4] =
            __builtin_convertvector(kt[i], bf16x4);
#pragma unroll
      for (int i = 0; i < 8; ++i) {
        sV[(fc + i * 4 + 0) * F_LD + fr] = (__bf16)vt[i].x;
        sV[(fc + i * 4 + 1) * F_LD + fr] = (__bf16)vt[i].y;
        sV[(fc + i * 4 + 2) * F_LD + fr] = (__bf16)vt[i].z;
        sV[(fc + i * 4 + 3) * F_LD + fr] = (__bf16)vt[i].w;
      }
    }
    __syncthreads();

    // S_j = Q Ktile^T for 4 column tiles of 16 kv each
    v8f sj[4];
#pragma unroll
    for (int j = 0; j < 4; ++j) {
      v16bf kf0 = ldfragB(&sK[(j * 16 + lrow) * F_LD + kb]);
      v16bf kf1 = ldfragB(&sK[(j * 16 + lrow) * F_LD + 32 + kb]);
      v8f sa = {};
      sa = wmma_bf16(qf0, kf0, sa);
      sa = wmma_bf16(qf1, kf1, sa);
      sj[j] = sa;
    }

    // scale + mask + row max (C layout: lane holds col j*16 + lane&15)
    float rmax[8];
#pragma unroll
    for (int r = 0; r < 8; ++r) rmax[r] = -3e38f;
#pragma unroll
    for (int j = 0; j < 4; ++j) {
      int mk = mrow_ptr[kv + j * 16 + (lane & 15)];
#pragma unroll
      for (int r = 0; r < 8; ++r) {
        float e = (mk != 0) ? sj[j][r] * scale : -1e10f;
        sj[j][r] = e;
        rmax[r] = fmaxf(rmax[r], e);
      }
    }
#pragma unroll
    for (int off = 1; off < 16; off <<= 1)
#pragma unroll
      for (int r = 0; r < 8; ++r)
        rmax[r] = fmaxf(rmax[r], __shfl_xor(rmax[r], off, 32));

    // online softmax update
    float alpha[8], rsum[8];
#pragma unroll
    for (int r = 0; r < 8; ++r) {
      float mn = fmaxf(mrun[r], rmax[r]);
      alpha[r] = __expf(mrun[r] - mn);
      mrun[r] = mn;
      rsum[r] = 0.0f;
    }
#pragma unroll
    for (int j = 0; j < 4; ++j)
#pragma unroll
      for (int r = 0; r < 8; ++r) {
        float p = __expf(sj[j][r] - mrun[r]);
        sj[j][r] = p;
        rsum[r] += p;
      }
#pragma unroll
    for (int off = 1; off < 16; off <<= 1)
#pragma unroll
      for (int r = 0; r < 8; ++r) rsum[r] += __shfl_xor(rsum[r], off, 32);
#pragma unroll
    for (int r = 0; r < 8; ++r) lrun[r] = lrun[r] * alpha[r] + rsum[r];
#pragma unroll
    for (int jj = 0; jj < 4; ++jj)
#pragma unroll
      for (int r = 0; r < 8; ++r) acc[jj][r] *= alpha[r];

    // spill P (C layout) to LDS as bf16, re-read as A fragments (wave-local,
    // DS ops are in-order per wave -> no barrier needed)
    __bf16* pw = sP[w];
#pragma unroll
    for (int j = 0; j < 4; ++j)
#pragma unroll
      for (int r = 0; r < 8; ++r)
        pw[(rbase + r) * F_LD + j * 16 + (lane & 15)] = (__bf16)sj[j][r];

    v16bf pf0 = ldfragA(&pw[lrow * F_LD + ka]);
    v16bf pf1 = ldfragA(&pw[lrow * F_LD + 32 + ka]);
#pragma unroll
    for (int jj = 0; jj < 4; ++jj) {
      v16bf vf0 = ldfragB(&sV[(jj * 16 + lrow) * F_LD + kb]);
      v16bf vf1 = ldfragB(&sV[(jj * 16 + lrow) * F_LD + 32 + kb]);
      acc[jj] = wmma_bf16(pf0, vf0, acc[jj]);
      acc[jj] = wmma_bf16(pf1, vf1, acc[jj]);
    }
  }

  // out = acc / l
  const int col = lane & 15;
#pragma unroll
  for (int jj = 0; jj < 4; ++jj)
#pragma unroll
    for (int r = 0; r < 8; ++r) {
      int gq = qb + w * 16 + rbase + r;
      o[rowstart + (size_t)gq * D + jj * 16 + col] = acc[jj][r] / lrun[r];
    }
}

// ---------------------------------------------------------------------------
// x[m,:] = LayerNorm(x[m,:] + res[m,:]) * g + be   (in place on x)
// grid: M blocks, 256 threads, D=512
// ---------------------------------------------------------------------------
__global__ __launch_bounds__(256) void ln_residual_kernel(
    float* __restrict__ x, const float* __restrict__ res,
    const float* __restrict__ g, const float* __restrict__ be) {
  const int D = 512;
  __shared__ float red[8];
  const int m = blockIdx.x, t = threadIdx.x;
  const int lane = t & 31, w = t >> 5;
  const size_t base = (size_t)m * D;

  float v0 = x[base + t] + res[base + t];
  float v1 = x[base + t + 256] + res[base + t + 256];

  float s = v0 + v1;
#pragma unroll
  for (int off = 16; off >= 1; off >>= 1) s += __shfl_xor(s, off, 32);
  if (lane == 0) red[w] = s;
  __syncthreads();
  float tot = 0.0f;
#pragma unroll
  for (int i = 0; i < 8; ++i) tot += red[i];
  float mu = tot * (1.0f / D);
  __syncthreads();

  float d0 = v0 - mu, d1 = v1 - mu;
  s = d0 * d0 + d1 * d1;
#pragma unroll
  for (int off = 16; off >= 1; off >>= 1) s += __shfl_xor(s, off, 32);
  if (lane == 0) red[w] = s;
  __syncthreads();
  tot = 0.0f;
#pragma unroll
  for (int i = 0; i < 8; ++i) tot += red[i];
  float rs = rsqrtf(tot * (1.0f / D) + 1e-5f);

  x[base + t] = d0 * rs * g[t] + be[t];
  x[base + t + 256] = d1 * rs * g[t + 256] + be[t + 256];
}

// ---------------------------------------------------------------------------
// Host orchestration
// ---------------------------------------------------------------------------
extern "C" void kernel_launch(void* const* d_in, const int* in_sizes, int n_in,
                              void* d_out, int out_size, void* d_ws,
                              size_t ws_size, hipStream_t stream) {
  (void)in_sizes; (void)n_in; (void)out_size; (void)ws_size;
  const int B = 4, S = 2048, D = 512, L = 6, H = 8, F = 2048;
  const int M = B * S;

  const int*   src    = (const int*)d_in[0];
  const int*   smask  = (const int*)d_in[1];
  const float* tokemb = (const float*)d_in[2];
  const float* Wq = (const float*)d_in[3];  const float* bq = (const float*)d_in[4];
  const float* Wk = (const float*)d_in[5];  const float* bk = (const float*)d_in[6];
  const float* Wv = (const float*)d_in[7];  const float* bv = (const float*)d_in[8];
  const float* Wo = (const float*)d_in[9];  const float* bo = (const float*)d_in[10];
  const float* W1 = (const float*)d_in[11]; const float* b1 = (const float*)d_in[12];
  const float* W2 = (const float*)d_in[13]; const float* b2 = (const float*)d_in[14];
  const float* g1 = (const float*)d_in[15]; const float* be1 = (const float*)d_in[16];
  const float* g2 = (const float*)d_in[17]; const float* be2 = (const float*)d_in[18];

  char* wp = (char*)d_ws;
  float* x  = (float*)wp; wp += (size_t)M * D * 4;
  float* qB = (float*)wp; wp += (size_t)M * D * 4;
  float* kB = (float*)wp; wp += (size_t)M * D * 4;
  float* vB = (float*)wp; wp += (size_t)M * D * 4;
  float* ao = (float*)wp; wp += (size_t)M * D * 4;
  float* pj = (float*)wp; wp += (size_t)M * D * 4;
  float* ff = (float*)wp; wp += (size_t)M * F * 4;
  __bf16* WqT = (__bf16*)wp; wp += (size_t)L * D * D * 2;
  __bf16* WkT = (__bf16*)wp; wp += (size_t)L * D * D * 2;
  __bf16* WvT = (__bf16*)wp; wp += (size_t)L * D * D * 2;
  __bf16* WoT = (__bf16*)wp; wp += (size_t)L * D * D * 2;
  __bf16* W1T = (__bf16*)wp; wp += (size_t)L * D * F * 2;
  __bf16* W2T = (__bf16*)wp; wp += (size_t)L * F * D * 2;

  // 1) weight transpose+convert (bf16 [N][K] per layer)
  dim3 tb(32, 8);
  convT_kernel<<<dim3(D / 32, D / 32, L), tb, 0, stream>>>(Wq, WqT, D, D);
  convT_kernel<<<dim3(D / 32, D / 32, L), tb, 0, stream>>>(Wk, WkT, D, D);
  convT_kernel<<<dim3(D / 32, D / 32, L), tb, 0, stream>>>(Wv, WvT, D, D);
  convT_kernel<<<dim3(D / 32, D / 32, L), tb, 0, stream>>>(Wo, WoT, D, D);
  convT_kernel<<<dim3(F / 32, D / 32, L), tb, 0, stream>>>(W1, W1T, D, F);
  convT_kernel<<<dim3(D / 32, F / 32, L), tb, 0, stream>>>(W2, W2T, F, D);

  // 2) embedding + positional encoding
  embed_pos_kernel<<<M, 256, 0, stream>>>(src, tokemb, x);

  // 3) layers
  dim3 gD(D / G_BN, M / G_BM);   // D-output GEMMs
  dim3 gF(F / G_BN, M / G_BM);   // F-output GEMM
  for (int l = 0; l < L; ++l) {
    const __bf16* wqt = WqT + (size_t)l * D * D;
    const __bf16* wkt = WkT + (size_t)l * D * D;
    const __bf16* wvt = WvT + (size_t)l * D * D;
    const __bf16* wot = WoT + (size_t)l * D * D;
    const __bf16* w1t = W1T + (size_t)l * D * F;
    const __bf16* w2t = W2T + (size_t)l * F * D;

    gemm_bf16_kernel<<<gD, 256, 0, stream>>>(x, wqt, bq + (size_t)l * D, qB, D, D, 0);
    gemm_bf16_kernel<<<gD, 256, 0, stream>>>(x, wkt, bk + (size_t)l * D, kB, D, D, 0);
    gemm_bf16_kernel<<<gD, 256, 0, stream>>>(x, wvt, bv + (size_t)l * D, vB, D, D, 0);

    flash_attn_kernel<<<dim3(S / 64, H, B), 128, 0, stream>>>(qB, kB, vB, smask, ao);

    gemm_bf16_kernel<<<gD, 256, 0, stream>>>(ao, wot, bo + (size_t)l * D, pj, D, D, 0);
    ln_residual_kernel<<<M, 256, 0, stream>>>(x, pj, g1 + (size_t)l * D, be1 + (size_t)l * D);

    gemm_bf16_kernel<<<gF, 256, 0, stream>>>(x, w1t, b1 + (size_t)l * F, ff, F, D, 1);
    gemm_bf16_kernel<<<gD, 256, 0, stream>>>(ff, w2t, b2 + (size_t)l * D, pj, D, F, 0);
    ln_residual_kernel<<<M, 256, 0, stream>>>(x, pj, g2 + (size_t)l * D, be2 + (size_t)l * D);
  }

  // 4) output
  hipMemcpyAsync(d_out, x, (size_t)M * D * sizeof(float),
                 hipMemcpyDeviceToDevice, stream);
}
